// MultiHeadAttention_7404523618567
// MI455X (gfx1250) — matrix operand order repaired and checked by reading, not compiled, not run on hardware
//
#include <hip/hip_runtime.h>
#include <stdint.h>

// ---------------------------------------------------------------------------
// Types / helpers
// ---------------------------------------------------------------------------
typedef unsigned short u16;
typedef __attribute__((ext_vector_type(16))) __bf16 v16bf;
typedef __attribute__((ext_vector_type(8)))  float  v8f;

union V16U { v16bf v; uint4 q[2]; u16 s[16]; };

static __device__ __forceinline__ v16bf load2x16(const u16* p0, const u16* p1) {
    V16U t;
    t.q[0] = *(const uint4*)p0;
    t.q[1] = *(const uint4*)p1;
    return t.v;
}
static __device__ __forceinline__ v16bf load32c(const u16* p) {
    V16U t;
    const uint4* q = (const uint4*)p;
    t.q[0] = q[0];
    t.q[1] = q[1];
    return t.v;
}
static __device__ __forceinline__ v8f wmma_bf16(v16bf a, v16bf b, v8f c) {
    return __builtin_amdgcn_wmma_f32_16x16x32_bf16(false, a, false, b,
                                                   (short)0, c, false, false);
}
static __device__ __forceinline__ u16 f2bf(float f) {
    unsigned u = __float_as_uint(f);
    unsigned r = u + 0x7fffu + ((u >> 16) & 1u);  // round-to-nearest-even
    return (u16)(r >> 16);
}
static __device__ __forceinline__ v8f zero8() {
    return (v8f){0.f, 0.f, 0.f, 0.f, 0.f, 0.f, 0.f, 0.f};
}
// Async global->LDS copy of 16B per lane (GLOBAL_LOAD_ASYNC_TO_LDS_B128,
// tracked by ASYNCcnt).  lds = byte offset within workgroup LDS allocation.
static __device__ __forceinline__ void async_ld16(unsigned lds,
                                                  const void* gp) {
    asm volatile("global_load_async_to_lds_b128 %0, %1, off"
                 :
                 : "v"(lds), "v"((unsigned long long)(uintptr_t)gp)
                 : "memory");
}
static __device__ __forceinline__ void wait_async0() {
    asm volatile("s_wait_asynccnt 0x0" ::: "memory");
}
static __device__ __forceinline__ void wait_async2() {
    asm volatile("s_wait_asynccnt 0x2" ::: "memory");
}

// ---------------------------------------------------------------------------
// fp32 -> bf16 conversion
// ---------------------------------------------------------------------------
__global__ void cvt_f32_bf16(const float* __restrict__ src,
                             u16* __restrict__ dst, int n) {
    int n4 = n >> 2;
    int stride = gridDim.x * blockDim.x;
    for (int i = blockIdx.x * blockDim.x + threadIdx.x; i < n4; i += stride) {
        float4 f = ((const float4*)src)[i];
        ushort4 o;
        o.x = f2bf(f.x); o.y = f2bf(f.y); o.z = f2bf(f.z); o.w = f2bf(f.w);
        ((ushort4*)dst)[i] = o;
    }
}

// ---------------------------------------------------------------------------
// C[M,N] = A[M,K] @ W[N,K]^T (+ bias), bf16 in, f32 accumulate.
// Block tile 128x128, 8 waves, each wave 32x64 (2 A-frags x 4 B-frags,
// 8 WMMA / k-step).  B tile TRIPLE-buffered in LDS via async-to-LDS copies:
// stage for phase j+2 issued at phase j; steady-state wait is
// s_wait_asynccnt<=2 (in-order completion retires exactly this phase's
// tile); only the peeled final phase drains to 0.
// ---------------------------------------------------------------------------
__global__ __launch_bounds__(256) void gemm_bf16_wmma(
    const u16* __restrict__ A, const u16* __restrict__ W,
    const float* __restrict__ bias, u16* __restrict__ outh,
    float* __restrict__ outf, int M, int N, int K) {
    constexpr int BSTR = 40;                 // padded k-stride (u16 units)
    __shared__ u16 Bsm[3][128 * BSTR];       // [n][k] B tiles, triple buffer

    const int tid  = threadIdx.x;
    const int lane = tid & 31;
    const int wid  = tid >> 5;
    const int wm   = wid >> 1;               // 0..3 : 32-row strip
    const int wn   = wid & 1;                // 0..1 : 64-col strip
    const int nblk = N / 128;
    const int gm0  = (blockIdx.x / nblk) * 128;
    const int gn0  = (blockIdx.x % nblk) * 128;
    const int lo = lane & 15, hi = lane >> 4;

    const u16* arow0 = A + (size_t)(gm0 + wm * 32 + lo) * K + hi * 8;
    const u16* arow1 = arow0 + (size_t)16 * K;

    // Hoisted staging addresses: this thread copies two fixed 16B chunks of
    // each 128x32 B tile (512 chunks total per tile).
    const int c0i = tid,        n0i = c0i >> 2, kc0 = c0i & 3;
    const int c1i = 256 + tid,  n1i = c1i >> 2, kc1 = c1i & 3;
    const u16* gW0 = W + (size_t)(gn0 + n0i) * K + kc0 * 8;
    const u16* gW1 = W + (size_t)(gn0 + n1i) * K + kc1 * 8;
    const unsigned off0 = (unsigned)(n0i * BSTR + kc0 * 8) * 2;
    const unsigned off1 = (unsigned)(n1i * BSTR + kc1 * 8) * 2;
    unsigned bufbase[3];
#pragma unroll
    for (int i = 0; i < 3; ++i)
        bufbase[i] = (unsigned)(uintptr_t)(&Bsm[i][0]);

    auto stageB = [&](int buf, int k0) {
        async_ld16(bufbase[buf] + off0, gW0 + k0);
        async_ld16(bufbase[buf] + off1, gW1 + k0);
    };

    v8f c[2][4];
#pragma unroll
    for (int mi = 0; mi < 2; ++mi)
#pragma unroll
        for (int nt = 0; nt < 4; ++nt) c[mi][nt] = zero8();

    // 2-stage pipelined LDS reads: load frag i+1/i+2 between WMMAs of frag i
    // so the scheduler can use counted s_wait_dscnt instead of full drains.
    auto compute = [&](int buf, v16bf a0, v16bf a1) {
        const u16* bb = &Bsm[buf][(wn * 64 + lo) * BSTR + hi * 8];
        v16bf b0 = load2x16(bb, bb + 16);
        v16bf b1 = load2x16(bb + 16 * BSTR, bb + 16 * BSTR + 16);
        c[0][0] = wmma_bf16(a0, b0, c[0][0]);
        c[1][0] = wmma_bf16(a1, b0, c[1][0]);
        v16bf b2 = load2x16(bb + 32 * BSTR, bb + 32 * BSTR + 16);
        c[0][1] = wmma_bf16(a0, b1, c[0][1]);
        c[1][1] = wmma_bf16(a1, b1, c[1][1]);
        v16bf b3 = load2x16(bb + 48 * BSTR, bb + 48 * BSTR + 16);
        c[0][2] = wmma_bf16(a0, b2, c[0][2]);
        c[1][2] = wmma_bf16(a1, b2, c[1][2]);
        c[0][3] = wmma_bf16(a0, b3, c[0][3]);
        c[1][3] = wmma_bf16(a1, b3, c[1][3]);
    };

    // Prologue: stage phases 0 and 1.
    stageB(0, 0);
    if (32 < K) stageB(1, 32);
    v16bf a0 = load2x16(arow0, arow0 + 16);
    v16bf a1 = load2x16(arow1, arow1 + 16);

    int buf = 0;
    int k0 = 0;
    for (; k0 < K - 32; k0 += 32) {          // steady state (phase P-1 peeled)
        wait_async2();                       // retire this phase's stage only
        __syncthreads();                     // tile visible to all waves
        if (k0 + 64 < K) {                   // stage phase j+2
            int b2i = buf + 2; if (b2i >= 3) b2i -= 3;
            stageB(b2i, k0 + 64);
        }
        v16bf a0n = load2x16(arow0 + k0 + 32, arow0 + k0 + 48);
        v16bf a1n = load2x16(arow1 + k0 + 32, arow1 + k0 + 48);
        compute(buf, a0, a1);
        a0 = a0n; a1 = a1n;
        buf = (buf == 2) ? 0 : buf + 1;
    }
    // Final phase: nothing left in flight behind it.
    wait_async0();
    __syncthreads();
    compute(buf, a0, a1);

#pragma unroll
    for (int mi = 0; mi < 2; ++mi)
#pragma unroll
        for (int nt = 0; nt < 4; ++nt) {
            const int ncol = gn0 + wn * 64 + nt * 16 + lo;
            const float bv = bias ? bias[ncol] : 0.f;
#pragma unroll
            for (int r = 0; r < 8; ++r) {
                const int row = gm0 + wm * 32 + mi * 16 + r + hi * 8;
                const float val = c[mi][nt][r] + bv;
                if (outf) outf[(size_t)row * N + ncol] = val;
                if (outh) outh[(size_t)row * N + ncol] = f2bf(val);
            }
        }
}

// ---------------------------------------------------------------------------
// Flash attention: one wave per 16-row q-tile, 4 waves/block.
// S = Q K^T via 4 WMMAs/16-col frag; streaming softmax in C-frag layout;
// P re-layout C->A frag via LDS; V staged transposed in LDS for PV B-frags.
// ---------------------------------------------------------------------------
__global__ __launch_bounds__(128) void flash_attn_kernel(
    const u16* __restrict__ qh, const u16* __restrict__ kh,
    const u16* __restrict__ vh, u16* __restrict__ ctxh) {
    constexpr int T = 2048, D = 2048, HD = 128, H = 16;
    constexpr int VSTRIDE = 40;  // padded k-stride for vT
    __shared__ u16 vT[4][HD * VSTRIDE];  // [d][k] transposed V tile per wave
    __shared__ u16 pT[4][16 * 32];       // [m][k] P tile per wave

    const int lane = threadIdx.x & 31;
    const int wid  = threadIdx.x >> 5;
    const int nqt  = T / 64;
    const int qt = blockIdx.x % nqt;
    const int h  = (blockIdx.x / nqt) % H;
    const int b  = blockIdx.x / (nqt * H);
    const int q0 = qt * 64 + wid * 16;
    const int lo = lane & 15, hi = lane >> 4;
    const size_t rowbase = (size_t)b * T;
    const float scale = 0.08838834764831845f;  // 1/sqrt(128)

    // Preload Q fragments (16 x 128 split into 4 x (16x32))
    v16bf qf[4];
    {
        const u16* qrow = qh + (rowbase + q0 + lo) * D + h * HD + hi * 8;
#pragma unroll
        for (int dc = 0; dc < 4; ++dc)
            qf[dc] = load2x16(qrow + dc * 32, qrow + dc * 32 + 16);
    }

    float mrow[8], lrow[8];
    v8f o[8];
#pragma unroll
    for (int r = 0; r < 8; ++r) { mrow[r] = -3e38f; lrow[r] = 0.f; }
#pragma unroll
    for (int dc = 0; dc < 8; ++dc) o[dc] = zero8();

    u16* vts = vT[wid];
    u16* pts = pT[wid];
    const int kend = q0 + 16;

    for (int k0 = 0; k0 < kend; k0 += 32) {
        // ---- S = scale * Q K^T for 32 key columns (2 frags of 16) ----
        v8f s[2];
#pragma unroll
        for (int f = 0; f < 2; ++f) {
            s[f] = zero8();
            const u16* krow =
                kh + (rowbase + k0 + f * 16 + lo) * D + h * HD + hi * 8;
#pragma unroll
            for (int dc = 0; dc < 4; ++dc)
                s[f] = wmma_bf16(qf[dc], load2x16(krow + dc * 32, krow + dc * 32 + 16), s[f]);
        }

        // ---- streaming softmax update (C-frag layout) ----
        float p0[8], p1[8];
#pragma unroll
        for (int r = 0; r < 8; ++r) {
            const int qg = q0 + r + 8 * hi;
            float a = s[0][r] * scale;
            float c = s[1][r] * scale;
            a = (k0 + lo      <= qg) ? a : -3e38f;
            c = (k0 + 16 + lo <= qg) ? c : -3e38f;
            float mx = fmaxf(a, c);
            mx = fmaxf(mx, __shfl_xor(mx, 1));
            mx = fmaxf(mx, __shfl_xor(mx, 2));
            mx = fmaxf(mx, __shfl_xor(mx, 4));
            mx = fmaxf(mx, __shfl_xor(mx, 8));
            const float mnew = fmaxf(mrow[r], mx);
            const float alpha = __expf(mrow[r] - mnew);
            const float e0 = __expf(a - mnew);
            const float e1 = __expf(c - mnew);
            float ls = e0 + e1;
            ls += __shfl_xor(ls, 1);
            ls += __shfl_xor(ls, 2);
            ls += __shfl_xor(ls, 4);
            ls += __shfl_xor(ls, 8);
            lrow[r] = lrow[r] * alpha + ls;
            mrow[r] = mnew;
            p0[r] = e0;
            p1[r] = e1;
#pragma unroll
            for (int dc = 0; dc < 8; ++dc) o[dc][r] *= alpha;
        }

        // ---- write P to LDS (C layout -> [m][k] array) ----
        asm volatile("s_wait_dscnt 0" ::: "memory");
#pragma unroll
        for (int r = 0; r < 8; ++r) {
            pts[(r + 8 * hi) * 32 + lo]      = f2bf(p0[r]);
            pts[(r + 8 * hi) * 32 + 16 + lo] = f2bf(p1[r]);
        }

        // ---- stage V tile transposed: vT[d][k] = V[k0+k][d] ----
        {
            const u16* vrow = vh + (rowbase + k0 + lane) * D + h * HD;
#pragma unroll
            for (int d8 = 0; d8 < 16; ++d8) {
                uint4 t = *(const uint4*)(vrow + d8 * 8);
                const u16* e = (const u16*)&t;
#pragma unroll
                for (int j = 0; j < 8; ++j)
                    vts[(d8 * 8 + j) * VSTRIDE + lane] = e[j];
            }
        }
        asm volatile("s_wait_dscnt 0" ::: "memory");

        // ---- O += P @ V ----
        v16bf pf = load2x16(pts + lo * 32 + hi * 8,
                            pts + lo * 32 + hi * 8 + 16);
#pragma unroll
        for (int dc = 0; dc < 8; ++dc) {
            v16bf bfv = load2x16(vts + (dc * 16 + lo) * VSTRIDE + hi * 8, vts + (dc * 16 + lo) * VSTRIDE + hi * 8 + 16);
            o[dc] = wmma_bf16(pf, bfv, o[dc]);
        }
    }

    // ---- epilogue: normalize and write bf16 context ----
    float inv[8];
#pragma unroll
    for (int r = 0; r < 8; ++r) inv[r] = 1.0f / lrow[r];
#pragma unroll
    for (int dc = 0; dc < 8; ++dc)
#pragma unroll
        for (int r = 0; r < 8; ++r)
            ctxh[(rowbase + q0 + r + 8 * hi) * D + h * HD + dc * 16 + lo] =
                f2bf(o[dc][r] * inv[r]);
}

// ---------------------------------------------------------------------------
// Launch
// ---------------------------------------------------------------------------
extern "C" void kernel_launch(void* const* d_in, const int* in_sizes, int n_in,
                              void* d_out, int out_size, void* d_ws,
                              size_t ws_size, hipStream_t stream) {
    constexpr int B = 4, T = 2048, D = 2048;
    constexpr size_t M = (size_t)B * T;        // 8192
    constexpr size_t TOK = M * D;              // 16,777,216 elems

    const float* x  = (const float*)d_in[0];
    const float* Wq = (const float*)d_in[1];
    const float* Wk = (const float*)d_in[2];
    const float* Wv = (const float*)d_in[3];
    const float* Wo = (const float*)d_in[4];
    const float* bo = (const float*)d_in[5];
    float* out = (float*)d_out;
    float* out_k = out + TOK;
    float* out_v = out + 2 * TOK;

    u16* ws = (u16*)d_ws;
    u16* xh  = ws;                       // 16,777,216 (also reused as ctx)
    u16* wqh = xh  + TOK;                // 4,194,304 each
    u16* wkh = wqh + (size_t)D * D;
    u16* wvh = wkh + (size_t)D * D;
    u16* woh = wvh + (size_t)D * D;
    u16* qhB = woh + (size_t)D * D;      // 16,777,216 each
    u16* khB = qhB + TOK;
    u16* vhB = khB + TOK;

    // 1) fp32 -> bf16
    cvt_f32_bf16<<<1024, 256, 0, stream>>>(x,  xh,  (int)TOK);
    cvt_f32_bf16<<<256,  256, 0, stream>>>(Wq, wqh, D * D);
    cvt_f32_bf16<<<256,  256, 0, stream>>>(Wk, wkh, D * D);
    cvt_f32_bf16<<<256,  256, 0, stream>>>(Wv, wvh, D * D);
    cvt_f32_bf16<<<256,  256, 0, stream>>>(Wo, woh, D * D);

    // 2) projections: q/k/v = x @ W^T  (k, v also to fp32 outputs)
    const int gemm_blocks = (int)((M / 128) * (D / 128));  // 1024
    gemm_bf16_wmma<<<gemm_blocks, 256, 0, stream>>>(xh, wqh, nullptr, qhB,
                                                    nullptr, (int)M, D, D);
    gemm_bf16_wmma<<<gemm_blocks, 256, 0, stream>>>(xh, wkh, nullptr, khB,
                                                    out_k, (int)M, D, D);
    gemm_bf16_wmma<<<gemm_blocks, 256, 0, stream>>>(xh, wvh, nullptr, vhB,
                                                    out_v, (int)M, D, D);

    // 3) causal flash attention -> ctx (reuses xh)
    flash_attn_kernel<<<B * 16 * (T / 64), 128, 0, stream>>>(qhB, khB, vhB, xh);

    // 4) out = ctx @ Wo^T + bo
    gemm_bf16_wmma<<<gemm_blocks, 256, 0, stream>>>(xh, woh, bo, nullptr, out,
                                                    (int)M, D, D);
}
